// Seq2Grid_1082331759187
// MI455X (gfx1250) — matrix operand order repaired and checked by reading, not compiled, not run on hardware
//
#include <hip/hip_runtime.h>
#include <hip/hip_bf16.h>

// ---------------------------------------------------------------------------
// CDNA5 (gfx1250, wave32) implementation.
// Precision: bf16 operands, f32 accumulate via V_WMMA_F32_16X16X32_BF16.
// Register blocking: gemm1 2x2 tiles/wave, gemm2 2x4 tiles/wave so each
// fragment load feeds 2 WMMAs (load bytes per WMMA: 32B / 24B).
// ---------------------------------------------------------------------------

typedef __bf16 bf16;
typedef __attribute__((ext_vector_type(16))) __bf16 v16bf;
typedef __attribute__((ext_vector_type(8)))  __bf16 v8bf;
typedef __attribute__((ext_vector_type(8)))  float  v8f;

#define B_  32
#define S_  2048
#define E_  512
#define U_  512

// ---- WMMA fragment loaders (layouts per CDNA5 ISA §7.12.2) -----
// A 16x32 bf16: lane (half=lane>>4, m=lane&15) holds row m,
//   elems 0..7  -> K = 8*half + e ; elems 8..15 -> K = 16 + 8*half + (e-8)
// => two contiguous 8-bf16 (16B) loads from the A row.
__device__ __forceinline__ v16bf load_a_frag(const bf16* rowk0, int half) {
    union { v16bf v; v8bf h[2]; } f;
    f.h[0] = *(const v8bf*)(rowk0 + 8 * half);
    f.h[1] = *(const v8bf*)(rowk0 + 16 + 8 * half);
    return f.v;
}
// B 32x16 bf16: lane (half, n=lane&15) holds column n, same K order as A:
//   elems 0..7 -> K = 8*half + e ; elems 8..15 -> K = 16 + 8*half + (e-8).
__device__ __forceinline__ v16bf load_bt_frag(const bf16* rowk0, int half) {
    union { v16bf v; v8bf h[2]; } f;
    f.h[0] = *(const v8bf*)(rowk0 + 8 * half);
    f.h[1] = *(const v8bf*)(rowk0 + 16 + 8 * half);
    return f.v;
}

__device__ __forceinline__ v8f wmma_bf16(v16bf a, v16bf b, v8f c) {
    return __builtin_amdgcn_wmma_f32_16x16x32_bf16(
        /*neg_a=*/false, a, /*neg_b=*/false, b,
        /*c_mod=*/(short)0, c, /*reuse_a=*/false, /*reuse_b=*/false);
}

__device__ __forceinline__ float fast_rcp(float x) {
    return __builtin_amdgcn_rcpf(x);   // v_rcp_f32
}

// ---------------------------------------------------------------------------
// Kernel 1: seq (f32, [B][S][E]) -> seq_bf ([B][S][E], bf16)
//                                   seqTrev ([B][E][S], bf16), s reversed:
//   seqTrev[b][e][s'] = seq[b][S-1-s'][e]
// 32x32 tiles, LDS transpose, all global accesses coalesced.
// ---------------------------------------------------------------------------
__global__ __launch_bounds__(256) void convert_seq_kernel(
    const float* __restrict__ seq, bf16* __restrict__ seq_bf,
    bf16* __restrict__ seqTrev) {
    __shared__ float tile[32][33];
    const int e0 = blockIdx.x * 32;
    const int s0 = blockIdx.y * 32;
    const int b  = blockIdx.z;
    const float* base = seq + ((size_t)b * S_ + s0) * E_ + e0;
#pragma unroll
    for (int it = 0; it < 4; ++it) {
        int idx = it * 256 + threadIdx.x;
        int i = idx >> 5, j = idx & 31;            // i: s-offset, j: e-offset
        float v = base[(size_t)i * E_ + j];
        tile[i][j] = v;
        seq_bf[((size_t)b * S_ + s0 + i) * E_ + e0 + j] = (bf16)v;
    }
    __syncthreads();
#pragma unroll
    for (int it = 0; it < 4; ++it) {
        int idx = it * 256 + threadIdx.x;
        int row = idx >> 5, col = idx & 31;        // row: e-offset, col: s'
        // s' = (S-1-s0-31) + col  <=>  s = s0 + 31 - col
        float v = tile[31 - col][row];             // stride-33 -> no conflicts
        seqTrev[((size_t)b * E_ + e0 + row) * S_ + (S_ - 1 - s0 - 31 + col)] =
            (bf16)v;
    }
}

__global__ __launch_bounds__(256) void convert_w1_kernel(
    const float* __restrict__ w1, bf16* __restrict__ w1bf) {
    int i = blockIdx.x * 256 + threadIdx.x;
    w1bf[i] = (bf16)w1[i];
}

// ---------------------------------------------------------------------------
// Kernel 2: fused GEMM1 + silu + GEMM2-epilogue + softmax.
// Wave handles a 32-row M-pair, N-tiles two at a time (2x2 WMMA block):
//   4 fragment loads -> 4 WMMAs.  h never hits memory; per-row logits are
// kept lane-local and reduced with shfl_xor inside each 16-lane half.
// ---------------------------------------------------------------------------
__global__ __launch_bounds__(256) void gemm1_actions_kernel(
    const bf16* __restrict__ Abf,   // [B*S][E]
    const bf16* __restrict__ w1bf,  // [U][E]  (== B^T rows)
    const float* __restrict__ b1, const float* __restrict__ w2,
    const float* __restrict__ b2, float* __restrict__ actions) {
    const int lane  = threadIdx.x & 31;
    const int wid   = threadIdx.x >> 5;
    const int mpair = blockIdx.x * 8 + wid;     // 32 rows per wave
    const int half  = lane >> 4;
    const int ln    = lane & 15;
    const int mbase = mpair * 32;

    const bf16* arow0 = Abf + (size_t)(mbase + ln) * E_;
    const bf16* arow1 = arow0 + (size_t)16 * E_;

    float l0[2][8], l1[2][8], l2[2][8];
#pragma unroll
    for (int mi = 0; mi < 2; ++mi)
#pragma unroll
        for (int r = 0; r < 8; ++r) {
            l0[mi][r] = 0.f; l1[mi][r] = 0.f; l2[mi][r] = 0.f;
        }

    for (int nt = 0; nt < U_ / 16; nt += 2) {
        const bf16* brow0 = w1bf + (size_t)(nt * 16 + ln) * E_;
        const bf16* brow1 = brow0 + (size_t)16 * E_;
        v8f acc[2][2] = {{v8f{}, v8f{}}, {v8f{}, v8f{}}};
#pragma unroll 4
        for (int kt = 0; kt < E_ / 32; ++kt) {
            v16bf a0 = load_a_frag(arow0 + kt * 32, half);
            v16bf a1 = load_a_frag(arow1 + kt * 32, half);
            v16bf bb0 = load_bt_frag(brow0 + kt * 32, half);
            v16bf bb1 = load_bt_frag(brow1 + kt * 32, half);
            acc[0][0] = wmma_bf16(a0, bb0, acc[0][0]);
            acc[0][1] = wmma_bf16(a0, bb1, acc[0][1]);
            acc[1][0] = wmma_bf16(a1, bb0, acc[1][0]);
            acc[1][1] = wmma_bf16(a1, bb1, acc[1][1]);
        }
#pragma unroll
        for (int ni = 0; ni < 2; ++ni) {
            const int u = (nt + ni) * 16 + ln;
            const float bias = b1[u];
            const float w20 = w2[u], w21 = w2[U_ + u], w22 = w2[2 * U_ + u];
#pragma unroll
            for (int mi = 0; mi < 2; ++mi) {
#pragma unroll
                for (int r = 0; r < 8; ++r) {
                    float x = acc[mi][ni][r] + bias;
                    float h = x * fast_rcp(1.f + __expf(-x));   // silu
                    l0[mi][r] += h * w20;
                    l1[mi][r] += h * w21;
                    l2[mi][r] += h * w22;
                }
            }
        }
    }
    // Reduce over the 16 lanes of each half (xor masks 1..8 stay in-half).
#pragma unroll
    for (int off = 1; off < 16; off <<= 1) {
#pragma unroll
        for (int mi = 0; mi < 2; ++mi)
#pragma unroll
            for (int r = 0; r < 8; ++r) {
                l0[mi][r] += __shfl_xor(l0[mi][r], off, 32);
                l1[mi][r] += __shfl_xor(l1[mi][r], off, 32);
                l2[mi][r] += __shfl_xor(l2[mi][r], off, 32);
            }
    }
    if (ln == 0) {  // lanes 0 and 16 write rows mbase + mi*16 + 8*half + r
#pragma unroll
        for (int mi = 0; mi < 2; ++mi)
#pragma unroll
            for (int r = 0; r < 8; ++r) {
                int row = mbase + mi * 16 + 8 * half + r;
                float x0 = l0[mi][r] + b2[0];
                float x1 = l1[mi][r] + b2[1];
                float x2 = l2[mi][r] + b2[2];
                float mx = fmaxf(x0, fmaxf(x1, x2));
                float e0 = __expf(x0 - mx), e1 = __expf(x1 - mx),
                      e2 = __expf(x2 - mx);
                float inv = fast_rcp(e0 + e1 + e2);
                actions[(size_t)row * 3 + 0] = e0 * inv;
                actions[(size_t)row * 3 + 1] = e1 * inv;
                actions[(size_t)row * 3 + 2] = e2 * inv;
            }
    }
}

// ---------------------------------------------------------------------------
// Kernel 3: sequential scan. One workgroup (256 threads) per batch; thread
// (h,w) owns T[h][w]. Double-buffered LDS state, one barrier per step.
// ws is staged in LDS 32 steps at a time and flushed TRANSPOSED:
//   wsT[b][hw][s] = ws_step_s[b][h][w]   (bf16)  -> A rows for kernel 4.
// ---------------------------------------------------------------------------
__global__ __launch_bounds__(256) void scan_kernel(
    const float* __restrict__ actions,  // [B][S][3]
    bf16* __restrict__ wsT) {           // [B][256][S]
    __shared__ float T[2][16][17];
    __shared__ float rs[2][16];
    __shared__ bf16  wst[256][32];
    const int b = blockIdx.x;
    const int t = threadIdx.x;
    const int h = t >> 4, w = t & 15;

    T[0][h][w] = (h == 0 && w == 15) ? 1.f : 0.f;
    if (w == 0) rs[0][h] = (h == 0) ? 1.f : 0.f;
    __syncthreads();

    int cur = 0;
    for (int s = 0; s < S_; ++s) {
        const float* act = actions + ((size_t)b * S_ + (S_ - 1 - s)) * 3;
        const float a0 = act[0], a1 = act[1], a2 = act[2];
        const float Tc = T[cur][h][w];
        wst[t][s & 31] = (bf16)((a0 + a1) * Tc);      // own row, no race
        float tnew;
        if (w < 15)
            tnew = a2 * Tc + a0 * T[cur][h][w + 1];
        else
            tnew = (h > 0 ? a1 * rs[cur][h - 1] : 0.f) + a2 * Tc;
        T[cur ^ 1][h][w] = tnew;
        if (w == 0)
            rs[cur ^ 1][h] =
                (a0 + a2) * rs[cur][h] + (h > 0 ? a1 * rs[cur][h - 1] : 0.f);
        cur ^= 1;
        if ((s & 31) == 31) {  // flush own 64B row (4x b128 stores)
            const uint4* src = (const uint4*)&wst[t][0];
            uint4* dst = (uint4*)(wsT + ((size_t)b * 256 + t) * S_ + (s - 31));
            dst[0] = src[0]; dst[1] = src[1]; dst[2] = src[2]; dst[3] = src[3];
        }
        __syncthreads();
    }
}

// ---------------------------------------------------------------------------
// Kernel 4: batched GEMM  G[b][hw][e] = sum_s wsT[b][hw][s] * seqTrev[b][e][s]
// M=256, N=512, K=2048 per batch.  Wave: 2 M-tiles x 4 N-tiles (2x4 block):
// 6 fragment loads -> 8 WMMAs per k-step.
// ---------------------------------------------------------------------------
__global__ __launch_bounds__(256) void gemm2_kernel(
    const bf16* __restrict__ wsT,      // [B][256][S]  A rows
    const bf16* __restrict__ seqTrev,  // [B][E][S]    B^T rows
    float* __restrict__ G) {           // [B][256][E]
    const int lane = threadIdx.x & 31;
    const int wid  = threadIdx.x >> 5;
    const int gw   = blockIdx.x * 8 + wid;
    const int b     = gw >> 6;        // 64 waves per batch
    const int rem   = gw & 63;
    const int mpair = rem >> 3;       // 0..7 -> mtiles 2*mpair, 2*mpair+1
    const int ngrp  = rem & 7;        // 4 ntiles each -> covers N=512
    const int half = lane >> 4, ln = lane & 15;

    const bf16* arow0 = wsT + ((size_t)b * 256 + mpair * 32 + ln) * S_;
    const bf16* arow1 = arow0 + (size_t)16 * S_;
    const bf16* brow0 = seqTrev + ((size_t)b * E_ + ngrp * 64 + ln) * S_;

    v8f acc[2][4] = {{v8f{}, v8f{}, v8f{}, v8f{}},
                     {v8f{}, v8f{}, v8f{}, v8f{}}};
#pragma unroll 2
    for (int kt = 0; kt < S_ / 32; ++kt) {
        v16bf a0 = load_a_frag(arow0 + kt * 32, half);
        v16bf a1 = load_a_frag(arow1 + kt * 32, half);
#pragma unroll
        for (int j = 0; j < 4; ++j) {
            v16bf bb =
                load_bt_frag(brow0 + (size_t)j * 16 * S_ + kt * 32, half);
            acc[0][j] = wmma_bf16(a0, bb, acc[0][j]);
            acc[1][j] = wmma_bf16(a1, bb, acc[1][j]);
        }
    }
#pragma unroll
    for (int mi = 0; mi < 2; ++mi) {
#pragma unroll
        for (int j = 0; j < 4; ++j) {
            const int n0 = (ngrp * 4 + j) * 16;
#pragma unroll
            for (int r = 0; r < 8; ++r) {
                int hw = mpair * 32 + mi * 16 + 8 * half + r;
                G[((size_t)b * 256 + hw) * E_ + n0 + ln] = acc[mi][j][r];
            }
        }
    }
}

// ---------------------------------------------------------------------------
// Workspace layout (bytes, all 256-aligned):
//   seq_bf   [B][S][E] bf16 :  67,108,864
//   seqTrev  [B][E][S] bf16 :  67,108,864
//   w1bf     [U][E]    bf16 :     524,288
//   actions  [B][S][3] f32  :     786,432
//   wsT      [B][256][S] bf16: 33,554,432
// total ~169.1 MB
// ---------------------------------------------------------------------------
extern "C" void kernel_launch(void* const* d_in, const int* in_sizes, int n_in,
                              void* d_out, int out_size, void* d_ws,
                              size_t ws_size, hipStream_t stream) {
    const float* seq = (const float*)d_in[0];
    const float* w1  = (const float*)d_in[1];
    const float* b1  = (const float*)d_in[2];
    const float* w2  = (const float*)d_in[3];
    const float* b2  = (const float*)d_in[4];

    char* wsb = (char*)d_ws;
    const size_t off_seqbf   = 0;
    const size_t off_seqTrev = off_seqbf + (size_t)B_ * S_ * E_ * 2;
    const size_t off_w1bf    = off_seqTrev + (size_t)B_ * S_ * E_ * 2;
    const size_t off_actions = off_w1bf + (size_t)U_ * E_ * 2;
    const size_t off_wsT     = off_actions + (size_t)B_ * S_ * 3 * 4;
    const size_t need        = off_wsT + (size_t)B_ * 256 * S_ * 2;
    if (ws_size < need) return;  // insufficient scratch

    bf16*  seq_bf  = (bf16*)(wsb + off_seqbf);
    bf16*  seqTrev = (bf16*)(wsb + off_seqTrev);
    bf16*  w1bf    = (bf16*)(wsb + off_w1bf);
    float* actions = (float*)(wsb + off_actions);
    bf16*  wsT     = (bf16*)(wsb + off_wsT);
    float* G       = (float*)d_out;

    convert_seq_kernel<<<dim3(E_ / 32, S_ / 32, B_), 256, 0, stream>>>(
        seq, seq_bf, seqTrev);
    convert_w1_kernel<<<(U_ * E_) / 256, 256, 0, stream>>>(w1, w1bf);
    gemm1_actions_kernel<<<(B_ * S_) / (32 * 8), 256, 0, stream>>>(
        seq_bf, w1bf, b1, w2, b2, actions);
    scan_kernel<<<B_, 256, 0, stream>>>(actions, wsT);
    gemm2_kernel<<<(B_ * 64) / 8, 256, 0, stream>>>(wsT, seqTrev, G);
}
